// TCNEncoder_40733469835945
// MI455X (gfx1250) — matrix operand order repaired and checked by reading, not compiled, not run on hardware
//
#include <hip/hip_runtime.h>
#include <hip/hip_bf16.h>
#include <math.h>

#define B_  128
#define T_  1024
#define F_  256
#define C_  128
#define NL_ 8
#define MT  4            // M-tiles (of 16 rows) per block in the layer kernel

typedef __attribute__((ext_vector_type(16))) __bf16        v16bf;
typedef __attribute__((ext_vector_type(8)))  float         v8f;
typedef __attribute__((ext_vector_type(8)))  unsigned int  v8u;
typedef __attribute__((ext_vector_type(4)))  int           v4i;

union Frag { v8u u; v16bf b; };

#if __has_builtin(__builtin_amdgcn_global_load_async_to_lds_b128)
#define ASYNC_CP 1
typedef __attribute__((address_space(1))) v4i g_v4i;
typedef __attribute__((address_space(3))) v4i l_v4i;
#else
#define ASYNC_CP 0
#endif

__device__ __forceinline__ void wait_async0() {
#if __has_builtin(__builtin_amdgcn_s_wait_asynccnt)
  __builtin_amdgcn_s_wait_asynccnt(0);
#elif ASYNC_CP
  asm volatile("s_wait_asynccnt 0x0" ::: "memory");
#endif
}

__device__ __forceinline__ unsigned short f2bf(float f) {
  __hip_bfloat16 h = __float2bfloat16(f);
  return __builtin_bit_cast(unsigned short, h);
}
__device__ __forceinline__ unsigned int pack2bf(float a, float b) {
  return ((unsigned int)f2bf(b) << 16) | (unsigned int)f2bf(a);
}
__device__ __forceinline__ float sigmoidf_(float x) {
  return 1.0f / (1.0f + __expf(-x));
}

// ---------------------------------------------------------------------------
// Weight conversion: Wh/Wc [F,C] f32 -> wenc [n=256][f=256] bf16 (B^T layout)
// ---------------------------------------------------------------------------
__global__ void convert_enc_w(const float* __restrict__ Wh,
                              const float* __restrict__ Wc,
                              unsigned short* __restrict__ wenc) {
  int i = blockIdx.x * blockDim.x + threadIdx.x;
  if (i >= 2 * C_ * F_) return;
  int n = i / F_;
  int f = i % F_;
  float v = (n < C_) ? Wh[f * C_ + n] : Wc[f * C_ + (n - C_)];
  wenc[n * F_ + f] = f2bf(v);
}

// conv_w [NL][4C][C][FW] f32 -> wconv [NL][tap][4C][C] bf16 ([N][K] per tap)
__global__ void convert_conv_w(const float* __restrict__ cw,
                               unsigned short* __restrict__ wconv) {
  int i = blockIdx.x * blockDim.x + threadIdx.x;
  if (i >= NL_ * 2 * 4 * C_ * C_) return;
  int ic  = i % C_;
  int o   = (i / C_) % (4 * C_);
  int tap = (i / (C_ * 4 * C_)) % 2;
  int l   = i / (C_ * 4 * C_ * 2);
  float v = cw[(((size_t)l * 4 * C_ + o) * C_ + ic) * 2 + tap];
  wconv[i] = f2bf(v);
}

// ---------------------------------------------------------------------------
// Encoder: h = tanh(x@Wh+bh) -> bf16 ; c = tanh(x@Wc+bc) -> f32
// ---------------------------------------------------------------------------
__global__ void __launch_bounds__(512) encoder_kernel(
    const float* __restrict__ x, const unsigned short* __restrict__ wenc,
    const float* __restrict__ bh, const float* __restrict__ bc,
    unsigned short* __restrict__ hout, float* __restrict__ cbuf) {
  __shared__ unsigned int xt[16 * 128];
  const int tid = threadIdx.x;
  const long row0 = (long)blockIdx.x * 16;

  {
    int r  = tid >> 5;
    int c0 = (tid & 31) * 8;
    const float* src = x + (row0 + r) * F_ + c0;
    float4 a = *(const float4*)(src);
    float4 b = *(const float4*)(src + 4);
    unsigned int* dst = &xt[r * 128 + (c0 >> 1)];
    dst[0] = pack2bf(a.x, a.y);
    dst[1] = pack2bf(a.z, a.w);
    dst[2] = pack2bf(b.x, b.y);
    dst[3] = pack2bf(b.z, b.w);
  }
  __syncthreads();

  const int wave = tid >> 5;
  const int lane = tid & 31;
  const int r    = lane & 15;
  const int hi   = lane >> 4;
  const int n    = wave * 16 + r;

  v8f acc = {};
  const unsigned int* wdw = (const unsigned int*)wenc;
#pragma unroll
  for (int ks = 0; ks < 8; ks++) {
    const int k0 = ks * 32;
    Frag A, Bm;
    const unsigned int* ap = &xt[r * 128 + (k0 >> 1) + hi * 4];
    A.u[0] = ap[0]; A.u[1] = ap[1]; A.u[2] = ap[2];  A.u[3] = ap[3];
    A.u[4] = ap[8]; A.u[5] = ap[9]; A.u[6] = ap[10]; A.u[7] = ap[11];
    const unsigned int* bp = wdw + (size_t)n * (F_ / 2) + (k0 >> 1) + hi * 4;
#pragma unroll
    for (int q = 0; q < 8; q++) Bm.u[q] = bp[q < 4 ? q : q + 4];
    acc = __builtin_amdgcn_wmma_f32_16x16x32_bf16(false, A.b, false, Bm.b,
                                                  (short)0, acc, false, false);
  }

  const float bias = (n < C_) ? bh[n] : bc[n - C_];
#pragma unroll
  for (int j = 0; j < 8; j++) {
    float v = tanhf(acc[j] + bias);
    long rowm = row0 + hi * 8 + j;
    if (n < C_) hout[rowm * C_ + n] = f2bf(v);
    else        cbuf[rowm * C_ + (n - C_)] = v;
  }
}

// ---------------------------------------------------------------------------
// One dilated-conv + gating layer. Block: 64 time rows (4 M-tiles) x 512 ch.
// out[t] = W0 @ h[t - d/2] + W1 @ h[t + d - d/2] + b   (zero outside [0,T))
// c = sig(ig)*c + sig(cg)*tanh(cf) ; h = sig(eg)*tanh(c)
// ---------------------------------------------------------------------------
union ShMem {
  unsigned int at[2][(16 * MT) * 64];  // per tap: 64 rows x 128 bf16 (32 KB)
  float outs[16 * 512];                // one M-tile of conv output  (32 KB)
};

__global__ void __launch_bounds__(256) layer_kernel(
    const unsigned short* __restrict__ hin, unsigned short* __restrict__ hout,
    float* __restrict__ cbuf, const unsigned short* __restrict__ wl,
    const float* __restrict__ bias, int d, int last,
    float* __restrict__ out_f32) {
  __shared__ ShMem sh;

  const int tid = threadIdx.x;
  const int tilesPerBatch = T_ / (16 * MT);        // 16
  const int bb = blockIdx.x / tilesPerBatch;
  const int t0 = (blockIdx.x % tilesPerBatch) * (16 * MT);
  const long rowbase = (long)bb * T_;

  const int off0 = -(d >> 1);
  const int off1 = d - (d >> 1);

  // ---- stage both shifted A tiles into LDS (async on gfx1250) ----
#pragma unroll
  for (int tap = 0; tap < 2; tap++) {
    const int off = tap ? off1 : off0;
#pragma unroll
    for (int it = 0; it < (16 * MT * 16) / 256; it++) {   // 4 iters
      int chunk = it * 256 + tid;
      int ro = chunk >> 4;                 // 0..63
      int cb = (chunk & 15) * 16;          // byte offset in 256B row
      int t  = t0 + ro + off;
      char* lp = (char*)&sh.at[tap][0] + ro * 256 + cb;
      if (t >= 0 && t < T_) {
        const char* gp = (const char*)hin + ((rowbase + t) * C_) * 2 + cb;
#if ASYNC_CP
        __builtin_amdgcn_global_load_async_to_lds_b128(
            (g_v4i*)gp, (l_v4i*)(unsigned int)(uintptr_t)lp, 0, 0);
#else
        *(uint4*)lp = *(const uint4*)gp;
#endif
      } else {
        *(uint4*)lp = make_uint4(0u, 0u, 0u, 0u);
      }
    }
  }
  wait_async0();
  __syncthreads();

  const int wave = tid >> 5;
  const int lane = tid & 31;
  const int r    = lane & 15;
  const int hi   = lane >> 4;

  v8f acc[MT][4] = {};
  const unsigned int* wdw = (const unsigned int*)wl;
#pragma unroll
  for (int tap = 0; tap < 2; tap++) {
    const unsigned int* wt = wdw + (size_t)tap * (4 * C_) * (C_ / 2);
#pragma unroll
    for (int ks = 0; ks < 4; ks++) {
      const int k0 = ks * 32;
      Frag A[MT];
#pragma unroll
      for (int m = 0; m < MT; m++) {
        const unsigned int* ap =
            &sh.at[tap][(m * 16 + r) * 64 + (k0 >> 1) + hi * 4];
        A[m].u[0] = ap[0]; A[m].u[1] = ap[1]; A[m].u[2] = ap[2];  A[m].u[3] = ap[3];
        A[m].u[4] = ap[8]; A[m].u[5] = ap[9]; A[m].u[6] = ap[10]; A[m].u[7] = ap[11];
      }
#pragma unroll
      for (int j = 0; j < 4; j++) {
        const int o = wave * 64 + j * 16 + r;
        Frag Bm;
        const unsigned int* bp = wt + (size_t)o * 64 + (k0 >> 1) + hi * 4;
#pragma unroll
        for (int q = 0; q < 8; q++) Bm.u[q] = bp[q < 4 ? q : q + 4];
#pragma unroll
        for (int m = 0; m < MT; m++) {
          acc[m][j] = __builtin_amdgcn_wmma_f32_16x16x32_bf16(
              false, A[m].b, false, Bm.b, (short)0, acc[m][j], false, false);
        }
      }
    }
  }
  __syncthreads();   // A tiles dead; LDS is reused for gate staging

  // ---- gates, one M-tile at a time through LDS ----
#pragma unroll
  for (int mt = 0; mt < MT; mt++) {
#pragma unroll
    for (int j = 0; j < 4; j++) {
      const int n = wave * 64 + j * 16 + r;
#pragma unroll
      for (int q = 0; q < 8; q++) {
        sh.outs[(hi * 8 + q) * 512 + n] = acc[mt][j][q];
      }
    }
    __syncthreads();
    {
      int m   = tid >> 4;
      int cc0 = (tid & 15) * 8;
      long row = rowbase + t0 + mt * 16 + m;
#pragma unroll
      for (int q = 0; q < 8; q++) {
        int cc = cc0 + q;
        float ig = sh.outs[m * 512 + cc]       + bias[cc];
        float cf = sh.outs[m * 512 + cc + 128] + bias[cc + 128];
        float cg = sh.outs[m * 512 + cc + 256] + bias[cc + 256];
        float eg = sh.outs[m * 512 + cc + 384] + bias[cc + 384];
        float cold = cbuf[row * C_ + cc];
        float cnew = sigmoidf_(ig) * cold + sigmoidf_(cg) * tanhf(cf);
        cbuf[row * C_ + cc] = cnew;
        float hn = sigmoidf_(eg) * tanhf(cnew);
        hout[row * C_ + cc] = f2bf(hn);
        if (last) out_f32[row * C_ + cc] = hn;
      }
    }
    __syncthreads();
  }
}

// ---------------------------------------------------------------------------
extern "C" void kernel_launch(void* const* d_in, const int* in_sizes, int n_in,
                              void* d_out, int out_size, void* d_ws, size_t ws_size,
                              hipStream_t stream) {
  (void)in_sizes; (void)n_in; (void)out_size; (void)ws_size;
  const float* x      = (const float*)d_in[0];
  const float* Wh     = (const float*)d_in[1];
  const float* bh     = (const float*)d_in[2];
  const float* Wc     = (const float*)d_in[3];
  const float* bc     = (const float*)d_in[4];
  const float* conv_w = (const float*)d_in[5];
  const float* conv_b = (const float*)d_in[6];

  char* ws = (char*)d_ws;
  const size_t HB = (size_t)B_ * T_ * C_;
  unsigned short* hbuf0 = (unsigned short*)(ws);
  unsigned short* hbuf1 = (unsigned short*)(ws + HB * 2);
  float*          cbuf  = (float*)(ws + HB * 4);
  unsigned short* wenc  = (unsigned short*)(ws + HB * 8);
  unsigned short* wconv = (unsigned short*)(ws + HB * 8 + (size_t)2 * C_ * F_ * 2);

  const int encw_n  = 2 * C_ * F_;
  const int convw_n = NL_ * 2 * 4 * C_ * C_;
  convert_enc_w<<<(encw_n + 255) / 256, 256, 0, stream>>>(Wh, Wc, wenc);
  convert_conv_w<<<(convw_n + 255) / 256, 256, 0, stream>>>(conv_w, wconv);

  encoder_kernel<<<(B_ * T_) / 16, 512, 0, stream>>>(x, wenc, bh, bc, hbuf0, cbuf);

  unsigned short* hin = hbuf0;
  unsigned short* ho  = hbuf1;
  for (int l = 0; l < NL_; l++) {
    layer_kernel<<<(B_ * T_) / (16 * MT), 256, 0, stream>>>(
        hin, ho, cbuf,
        wconv + (size_t)l * 2 * 4 * C_ * C_,
        conv_b + (size_t)l * 4 * C_,
        1 << l, (l == NL_ - 1) ? 1 : 0, (float*)d_out);
    unsigned short* tmp = hin; hin = ho; ho = tmp;
  }
}
